// GradientDescentMemory_12326556139948
// MI455X (gfx1250) — matrix operand order repaired and checked by reading, not compiled, not run on hardware
//
#include <hip/hip_runtime.h>

// Problem constants (reference: L=65536, D_K=D_V=128)
#define L_TOTAL 65536
#define D 128
#define NWG1 256                       // phase-1 workgroups (partial-J producers)
#define ROWS_PER_WG (L_TOTAL / NWG1)   // 256 rows of K/V per workgroup
#define CHUNK 64                       // rows staged in LDS per iteration
#define PAD 8                          // bf16 padding -> row stride 144B, kills bank conflicts
#define LSTRIDE (CHUNK + PAD)          // 72
#define JSTRIDE (D + PAD)              // 136

typedef __attribute__((ext_vector_type(16))) __bf16 v16bf;
typedef __attribute__((ext_vector_type(8)))  __bf16 v8bf;
typedef __attribute__((ext_vector_type(8)))  float  v8f;

__device__ __forceinline__ v8f zero8() {
  v8f z = {0.f, 0.f, 0.f, 0.f, 0.f, 0.f, 0.f, 0.f};
  return z;
}

__device__ __forceinline__ v16bf combine(v8bf lo, v8bf hi) {
  v16bf r;
#pragma unroll
  for (int i = 0; i < 8; ++i) { r[i] = lo[i]; r[i + 8] = hi[i]; }
  return r;
}

// ---------------------------------------------------------------------------
// Phase 1: partial_J[wg] = sum over this WG's L-slice of  (w*V)^T @ K
// A (16x32 bf16) = (w*V)^T tile  [M=v, K=l] ; B (32x16 bf16) = K tile [K=l, N=k]
// ---------------------------------------------------------------------------
__global__ __launch_bounds__(256)
void wj_phase1(const float* __restrict__ Kg, const float* __restrict__ Vg,
               const float* __restrict__ wg, float* __restrict__ partial) {
  __shared__ __align__(16) __bf16 sV[D][LSTRIDE];  // (w*V)^T : [v][l_local]
  __shared__ __align__(16) __bf16 sK[D][LSTRIDE];  // K^T     : [k][l_local]

  const int tid  = threadIdx.x;
  const int wave = tid >> 5;
  const int lane = tid & 31;
  const int hb   = lane >> 4;   // half-wave select
  const int m    = lane & 15;

  v8f acc[8];
#pragma unroll
  for (int n = 0; n < 8; ++n) acc[n] = zero8();

  const int base_l = blockIdx.x * ROWS_PER_WG;

  for (int sc = 0; sc < ROWS_PER_WG; sc += CHUNK) {
    const int gl0 = base_l + sc;
    __syncthreads();
    // Stage CHUNK x 128 of V (weighted) and K into LDS, transposed, as bf16.
    for (int i = tid; i < CHUNK * D / 4; i += 256) {
      const int ll = i >> 5;            // local l (0..CHUNK-1)
      const int d0 = (i & 31) * 4;      // d offset
      const int gl = gl0 + ll;
      const float  wl = wg[gl];
      const float4 v4 = *(const float4*)(Vg + (size_t)gl * D + d0);
      const float4 k4 = *(const float4*)(Kg + (size_t)gl * D + d0);
      sV[d0 + 0][ll] = (__bf16)(v4.x * wl);
      sV[d0 + 1][ll] = (__bf16)(v4.y * wl);
      sV[d0 + 2][ll] = (__bf16)(v4.z * wl);
      sV[d0 + 3][ll] = (__bf16)(v4.w * wl);
      sK[d0 + 0][ll] = (__bf16)k4.x;
      sK[d0 + 1][ll] = (__bf16)k4.y;
      sK[d0 + 2][ll] = (__bf16)k4.z;
      sK[d0 + 3][ll] = (__bf16)k4.w;
    }
    if (gl0 + CHUNK < L_TOTAL) {        // speculative prefetch of next slice
      __builtin_prefetch(Vg + (size_t)(gl0 + CHUNK) * D + tid * 4, 0, 0);
      __builtin_prefetch(Kg + (size_t)(gl0 + CHUNK) * D + tid * 4, 0, 0);
    }
    __syncthreads();

#pragma unroll
    for (int kk = 0; kk < CHUNK; kk += 32) {
      // A fragment (16x32): lanes0-15 elems0-7=K0..7, 8-15=K16..23;
      //                     lanes16-31 elems0-7=K8..15, 8-15=K24..31
      const int arow = wave * 16 + m;
      v16bf afrag = combine(*(const v8bf*)&sV[arow][kk + 8 * hb],
                            *(const v8bf*)&sV[arow][kk + 16 + 8 * hb]);
      // Prefetch ALL B fragments first so the 16 ds_load_b128 are in flight
      // together, then issue the 8 WMMAs back-to-back (XDL-friendly).
      v16bf bfrag[8];
#pragma unroll
      for (int n = 0; n < 8; ++n) {
        // B fragment (32x16): same K order as A (elems0-7=K8hb.., 8-15=K16+8hb..)
        const int brow = n * 16 + m;   // N column index -> row of K^T
        bfrag[n] = combine(*(const v8bf*)&sK[brow][kk + 8 * hb],
                           *(const v8bf*)&sK[brow][kk + 16 + 8 * hb]);
      }
#pragma unroll
      for (int n = 0; n < 8; ++n) {
        acc[n] = __builtin_amdgcn_wmma_f32_16x16x32_bf16(
            false, afrag, false, bfrag[n], (short)0, acc[n], false, false);
      }
    }
  }

  // C layout: VGPR r -> M = r + 8*hb within tile, N = m
  float* pout = partial + (size_t)blockIdx.x * (D * D);
#pragma unroll
  for (int n = 0; n < 8; ++n) {
#pragma unroll
    for (int r = 0; r < 8; ++r) {
      const int row = wave * 16 + r + 8 * hb;  // v
      const int col = n * 16 + m;              // k
      pout[row * D + col] = acc[n][r];
    }
  }
}

// ---------------------------------------------------------------------------
// Reduce the NWG1 partial J's (fixed order -> deterministic); emit J^T in bf16
// ---------------------------------------------------------------------------
__global__ __launch_bounds__(256)
void wj_reduce(const float* __restrict__ partial, unsigned short* __restrict__ JtRaw) {
  const int idx = blockIdx.x * 256 + threadIdx.x;  // = v*128 + kcol
  float s = 0.f;
  for (int p = 0; p < NWG1; ++p) s += partial[(size_t)p * (D * D) + idx];
  const int v    = idx >> 7;
  const int kcol = idx & 127;
  ((__bf16*)JtRaw)[kcol * D + v] = (__bf16)s;     // store transposed
}

// ---------------------------------------------------------------------------
// Phase 2: out = q @ J  (A = q tile from global w/ cvt, B = J from LDS J^T)
// ---------------------------------------------------------------------------
__global__ __launch_bounds__(256)
void wj_phase2(const float* __restrict__ qg, const unsigned short* __restrict__ JtRaw,
               float* __restrict__ out) {
  __shared__ __align__(16) __bf16 sJ[D][JSTRIDE];  // J^T : [kcol][v]

  const int tid  = threadIdx.x;
  const int wave = tid >> 5;
  const int lane = tid & 31;
  const int hb   = lane >> 4;
  const int m    = lane & 15;

  const v8bf* Jg = (const v8bf*)JtRaw;
  for (int i = tid; i < D * D / 8; i += 256) {
    const int row  = i >> 4;
    const int col0 = (i & 15) * 8;
    *(v8bf*)&sJ[row][col0] = Jg[i];
  }
  __syncthreads();

  const int lrow = blockIdx.x * 128 + wave * 16 + m;  // this lane's q row
  const float* qrow = qg + (size_t)lrow * D;

  v8f acc[8];
#pragma unroll
  for (int n = 0; n < 8; ++n) acc[n] = zero8();

#pragma unroll
  for (int v0 = 0; v0 < D; v0 += 32) {
    // Build A fragment directly from global f32 q (coalesced b128 loads + cvt)
    const float4 a0 = *(const float4*)(qrow + v0 + 8 * hb);
    const float4 a1 = *(const float4*)(qrow + v0 + 8 * hb + 4);
    const float4 a2 = *(const float4*)(qrow + v0 + 16 + 8 * hb);
    const float4 a3 = *(const float4*)(qrow + v0 + 16 + 8 * hb + 4);
    v16bf afrag;
    afrag[0]  = (__bf16)a0.x; afrag[1]  = (__bf16)a0.y;
    afrag[2]  = (__bf16)a0.z; afrag[3]  = (__bf16)a0.w;
    afrag[4]  = (__bf16)a1.x; afrag[5]  = (__bf16)a1.y;
    afrag[6]  = (__bf16)a1.z; afrag[7]  = (__bf16)a1.w;
    afrag[8]  = (__bf16)a2.x; afrag[9]  = (__bf16)a2.y;
    afrag[10] = (__bf16)a2.z; afrag[11] = (__bf16)a2.w;
    afrag[12] = (__bf16)a3.x; afrag[13] = (__bf16)a3.y;
    afrag[14] = (__bf16)a3.z; afrag[15] = (__bf16)a3.w;
    // Prefetch all B fragments, then burst the WMMAs.
    v16bf bfrag[8];
#pragma unroll
    for (int n = 0; n < 8; ++n) {
      const int brow = n * 16 + m;  // output column -> row of J^T
      bfrag[n] = combine(*(const v8bf*)&sJ[brow][v0 + 8 * hb],
                         *(const v8bf*)&sJ[brow][v0 + 16 + 8 * hb]);
    }
#pragma unroll
    for (int n = 0; n < 8; ++n) {
      acc[n] = __builtin_amdgcn_wmma_f32_16x16x32_bf16(
          false, afrag, false, bfrag[n], (short)0, acc[n], false, false);
    }
  }

  const int obase = blockIdx.x * 128 + wave * 16;
#pragma unroll
  for (int n = 0; n < 8; ++n) {
#pragma unroll
    for (int r = 0; r < 8; ++r) {
      out[(size_t)(obase + r + 8 * hb) * D + n * 16 + m] = acc[n][r];
    }
  }
}

// ---------------------------------------------------------------------------
extern "C" void kernel_launch(void* const* d_in, const int* in_sizes, int n_in,
                              void* d_out, int out_size, void* d_ws, size_t ws_size,
                              hipStream_t stream) {
  const float* q = (const float*)d_in[0];
  const float* K = (const float*)d_in[1];
  const float* V = (const float*)d_in[2];
  const float* w = (const float*)d_in[3];
  float* out = (float*)d_out;

  // ws layout: [0, NWG1*128*128 f32) partial J's; then 128*128 bf16 J^T
  float* partial = (float*)d_ws;
  unsigned short* Jt =
      (unsigned short*)((char*)d_ws + (size_t)NWG1 * D * D * sizeof(float));

  wj_phase1<<<NWG1, 256, 0, stream>>>(K, V, w, partial);
  wj_reduce<<<(D * D) / 256, 256, 0, stream>>>(partial, Jt);
  wj_phase2<<<L_TOTAL / 128, 256, 0, stream>>>(q, Jt, out);
}
